// VoxelEncoder_26396869001383
// MI455X (gfx1250) — matrix operand order repaired and checked by reading, not compiled, not run on hardware
//
#include <hip/hip_runtime.h>

typedef __attribute__((ext_vector_type(16))) _Float16 v16h;
typedef __attribute__((ext_vector_type(8)))  _Float16 v8h;
typedef __attribute__((ext_vector_type(8)))  float    v8f;

#define NVX    32
#define NBATCH 4
#define NPTS   32768   // points per batch
#define NVOX   32768   // 32^3

// ---------------------------------------------------------------------------
// A-fragment loader for v_wmma_f32_16x16x32_f16.
// Lane L (half = L/16, row = L%16) holds K in {half*16..+15}
// of row M = row. rowbase points at the 32-f16 K window of that row.
// ---------------------------------------------------------------------------
__device__ __forceinline__ v16h load_a_frag(const _Float16* rowbase, int half) {
  v8h c0 = *(const v8h*)(rowbase + half * 16);
  v8h c1 = *(const v8h*)(rowbase + half * 16 + 8);
  v16h a;
#pragma unroll
  for (int i = 0; i < 8; ++i) { a[i] = c0[i]; a[i + 8] = c1[i]; }
  return a;
}

// Async 16B global -> LDS copy (CDNA5). dst must be an LDS (shared) pointer,
// src a global pointer. Tracked by ASYNCcnt; complete with async_wait0().
__device__ __forceinline__ void async_copy_b128(void* lds_dst, const void* gsrc) {
  unsigned lds_addr = (unsigned)(size_t)lds_dst;   // generic->LDS: low 32 bits
  asm volatile("global_load_async_to_lds_b128 %0, %1, off"
               :: "v"(lds_addr), "v"(gsrc) : "memory");
}
__device__ __forceinline__ void async_wait0() {
  asm volatile("s_wait_asynccnt 0x0" ::: "memory");
}

// ---------------------------------------------------------------------------
// Utility kernels
// ---------------------------------------------------------------------------
__global__ void zero_u32(unsigned int* __restrict__ p, long n) {
  long i = (long)blockIdx.x * blockDim.x + threadIdx.x;
  if (i < n) p[i] = 0u;
}

__global__ void f32_to_f16(const float* __restrict__ in, _Float16* __restrict__ out, long n) {
  long stride = (long)gridDim.x * blockDim.x;
  for (long i = (long)blockIdx.x * blockDim.x + threadIdx.x; i < n; i += stride)
    out[i] = (_Float16)in[i];
}

// vox bits (f32, already channel-major NCDHW) -> f16, fully coalesced
__global__ void vox_to_act(const unsigned int* __restrict__ vox, _Float16* __restrict__ act) {
  long id = (long)blockIdx.x * blockDim.x + threadIdx.x;
  if (id >= (long)NBATCH * 256 * NVOX) return;
  act[id] = (_Float16)__uint_as_float(vox[id]);
}

// ---------------------------------------------------------------------------
// Point MLP (9->128->256, f16 WMMA) + scatter-max into voxel grid.
// Block = 256 threads = 8 waves; each wave owns 16 points (128 pts / block).
// Scatter writes the CHANNEL-MAJOR layout vox[b][c][voxel] so the downstream
// conv input needs no transpose (atomic targets are random either way).
// ---------------------------------------------------------------------------
__global__ __launch_bounds__(256)
void point_mlp_scatter(const float* __restrict__ pc,   // [B*N][6]
                       const float* __restrict__ w0,   // [128][9]
                       const float* __restrict__ b0,   // [128]
                       const float* __restrict__ w1,   // [256][128]
                       const float* __restrict__ b1,   // [256]
                       unsigned int* __restrict__ vox) // [B][256][NVOX]
{
  __shared__ __align__(64) _Float16 xls[128][32];    // padded point features
  __shared__ __align__(64) _Float16 h0ls[128][128];  // layer-0 activations
  __shared__ __align__(64) _Float16 w0ls[128][32];   // w0 zero-padded K 9->32
  __shared__ __align__(64) _Float16 w1ls[16][128];   // streamed w1 tile
  __shared__ int   flatls[128];
  __shared__ float b0ls[128];
  __shared__ float b1ls[256];

  const int tid  = threadIdx.x;
  const int lane = tid & 31;
  const int wave = tid >> 5;
  const int half = lane >> 4;
  const int l16  = lane & 15;
  const int p0   = blockIdx.x * 128;
  const int bidx = p0 / NPTS;   // all 128 points of a block share the batch idx

  // --- stage point features + w0 (once) ---
  if (tid < 128) {
    int gp = p0 + tid;
    float px = pc[gp * 6 + 0], py = pc[gp * 6 + 1], pz = pc[gp * 6 + 2];
    float f3 = pc[gp * 6 + 3], f4 = pc[gp * 6 + 4], f5 = pc[gp * 6 + 5];
    const float vsz = 1.0f / NVX;
    int ix = (int)floorf(px * NVX), iy = (int)floorf(py * NVX), iz = (int)floorf(pz * NVX);
    float cx = ((float)ix + 0.5f) * vsz, cy = ((float)iy + 0.5f) * vsz, cz = ((float)iz + 0.5f) * vsz;
    float f[9] = { px - cx, py - cy, pz - cz, cx, cy, cz, f3, f4, f5 };
#pragma unroll
    for (int j = 0; j < 9; ++j) xls[tid][j] = (_Float16)f[j];
#pragma unroll
    for (int j = 9; j < 32; ++j) xls[tid][j] = (_Float16)0.0f;
    flatls[tid] = (ix * NVX + iy) * NVX + iz;
    b0ls[tid] = b0[tid];
  }
  b1ls[tid] = b1[tid];
#pragma unroll
  for (int e = 0; e < 16; ++e) {          // 128*32 slots, 16 per thread
    int slot = tid * 16 + e;
    int n = slot >> 5, k = slot & 31;
    w0ls[n][k] = (k < 9) ? (_Float16)w0[n * 9 + k] : (_Float16)0.0f;
  }
  __syncthreads();

  // ---- layer 0: K = 32 (9 valid), 8 N-tiles of 16 -> 128 channels ----
  {
    v16h a = load_a_frag(&xls[wave * 16 + l16][0], half);
#pragma unroll 1
    for (int nt = 0; nt < 8; ++nt) {
      int n = nt * 16 + l16;
      v16h bf = *(const v16h*)&w0ls[n][half * 16];
      v8f acc = {};
      acc = __builtin_amdgcn_wmma_f32_16x16x32_f16(false, a, false, bf, (short)0, acc, false, false);
      float bs = b0ls[n];
#pragma unroll
      for (int i = 0; i < 8; ++i) {
        float v = acc[i] + bs; v = v > 0.0f ? v : 0.0f;
        h0ls[wave * 16 + i + 8 * half][n] = (_Float16)v;
      }
    }
  }
  __syncthreads();

  // ---- layer 1: K = 128 (4 steps), 16 N-tiles -> 256 channels; scatter ----
#pragma unroll 1
  for (int nt = 0; nt < 16; ++nt) {
    // cooperatively stage w1 tile [16][128] (f32 -> f16), coalesced 8/thread
    {
      int idx = tid * 8;
      int n = idx >> 7, k = idx & 127;
      const float* src = w1 + (long)(nt * 16 + n) * 128 + k;
      v8h pk;
#pragma unroll
      for (int i = 0; i < 8; ++i) pk[i] = (_Float16)src[i];
      *(v8h*)&w1ls[n][k] = pk;
    }
    __syncthreads();

    int n = nt * 16 + l16;
    v8f acc = {};
#pragma unroll
    for (int s = 0; s < 4; ++s) {
      int kb = s * 32;
      v16h a  = load_a_frag(&h0ls[wave * 16 + l16][kb], half);
      v16h bf = *(const v16h*)&w1ls[l16][kb + half * 16];
      acc = __builtin_amdgcn_wmma_f32_16x16x32_f16(false, a, false, bf, (short)0, acc, false, false);
    }
    float bs = b1ls[n];
#pragma unroll
    for (int i = 0; i < 8; ++i) {
      float v = acc[i] + bs; v = v > 0.0f ? v : 0.0f;
      int pl = wave * 16 + i + 8 * half;
      // channel-major: vox[b][n][flat]
      unsigned int* addr = vox + ((long)bidx * 256 + n) * NVOX + flatls[pl];
      atomicMax(addr, __float_as_uint(v));   // valid: v >= 0 after relu
    }
    __syncthreads();   // protect w1ls for next iteration
  }
}

// ---------------------------------------------------------------------------
// Implicit-GEMM 3x3x3 conv, 'SAME' pad, fused bias+ReLU. f16 in/out.
// M = B*D^3 (x fastest), N = COUT, K = CIN*27, k = ((cin*3+dz)*3+dy)*3+dx.
// Block = 256 threads = 8 waves -> 32(M) x 256(N) tile, K-step 64.
// Double-buffered LDS: A staged synchronously (predicated gather), B staged
// with global_load_async_to_lds_b128 (ASYNCcnt). 8 WMMA per wave per step.
// Epilogue packs each accumulator's 8 contiguous spatial outputs into one
// 16B store.
// ---------------------------------------------------------------------------
template <int CIN, int COUT, int D>
__global__ __launch_bounds__(256)
void conv3d_wmma(const _Float16* __restrict__ act,  // [B][CIN][D][D][D]
                 const _Float16* __restrict__ wh,   // [COUT][CIN*27] f16
                 const float* __restrict__ bias,    // [COUT]
                 _Float16* __restrict__ out)        // [B][COUT][D][D][D]
{
  constexpr int K = CIN * 27;
  constexpr int D3 = D * D * D;
  constexpr int AP = 72;  // padded A row (f16): 144B keeps 16B align, spreads banks
  __shared__ __align__(64) _Float16 Als[2][32][AP];
  __shared__ __align__(64) _Float16 Bls[2][256][64];

  const int tid  = threadIdx.x;
  const int lane = tid & 31;
  const int wave = tid >> 5;
  const int half = lane >> 4;
  const int l16  = lane & 15;
  const int mh   = wave >> 2;        // 0/1: which 16-row half
  const int nq   = wave & 3;         // which group of 4 n-subtiles
  const int m0   = blockIdx.x * 32;
  const int n0   = blockIdx.y * 256;

  // stage one 64-wide K slice into buffer `buf`
  auto stage = [&](int k0, int buf) {
    // A tile: 32 rows x 64 K; lane-major over rows => contiguous global runs
#pragma unroll
    for (int e = 0; e < 8; ++e) {
      int idx = tid + e * 256;
      int r = idx & 31;          // row
      int t = idx >> 5;          // k within step (0..63)
      int k = k0 + t;
      int cin = k / 27;
      int rem = k - cin * 27;
      int dz = rem / 9;
      int r2 = rem - dz * 9;
      int dy = r2 / 3;
      int dx = r2 - dy * 3;
      int m = m0 + r;
      int b = m / D3;
      int s = m - b * D3;
      int z = s / (D * D);
      int s2 = s - z * (D * D);
      int y = s2 / D;
      int x = s2 - y * D;
      int iz = z + dz - 1, iy = y + dy - 1, ixx = x + dx - 1;
      _Float16 v = (_Float16)0.0f;
      if (iz >= 0 && iz < D && iy >= 0 && iy < D && ixx >= 0 && ixx < D)
        v = act[(((long)b * CIN + cin) * D + iz) * (D * D) + iy * D + ixx];
      Als[buf][r][t] = v;
    }
    // B tile: 256 rows x 64 K = 2048 16B-chunks, 8 async copies per thread
#pragma unroll
    for (int j = 0; j < 8; ++j) {
      int cid = j * 256 + tid;
      int row = cid >> 3, seg = cid & 7;
      async_copy_b128(&Bls[buf][row][seg * 8],
                      wh + (long)(n0 + row) * K + k0 + seg * 8);
    }
  };

  v8f acc[4] = {};
  stage(0, 0);
  async_wait0();
  __syncthreads();

  constexpr int NSTEP = K / 64;
  for (int step = 0; step < NSTEP; ++step) {
    int cur = step & 1;
    if (step + 1 < NSTEP) stage((step + 1) * 64, cur ^ 1);

    // 2 K-slices of 32, 4 n-subtiles each: 8 WMMA on the current buffer
#pragma unroll
    for (int s = 0; s < 2; ++s) {
      v16h a = load_a_frag(&Als[cur][mh * 16 + l16][s * 32], half);
#pragma unroll
      for (int j = 0; j < 4; ++j) {
        v16h bf = *(const v16h*)&Bls[cur][(nq * 4 + j) * 16 + l16][s * 32 + half * 16];
        acc[j] = __builtin_amdgcn_wmma_f32_16x16x32_f16(false, a, false, bf, (short)0,
                                                        acc[j], false, false);
      }
    }
    async_wait0();     // next buffer's async B copies landed
    __syncthreads();   // all waves done reading `cur` and staging `cur^1`
  }

  // epilogue: bias + relu; lane's 8 C-rows are 8 contiguous spatial outputs
  // (base multiple of 8, D3 multiple of 8 => never crosses batch boundary)
  {
    int mbase = m0 + mh * 16 + 8 * half;
    int b = mbase / D3;
    int sbase = mbase - b * D3;
#pragma unroll
    for (int j = 0; j < 4; ++j) {
      int n = n0 + (nq * 4 + j) * 16 + l16;
      float bs = bias[n];
      v8h pk;
#pragma unroll
      for (int i = 0; i < 8; ++i) {
        float v = acc[j][i] + bs;
        v = v > 0.0f ? v : 0.0f;
        pk[i] = (_Float16)v;
      }
      *(v8h*)(out + ((long)b * COUT + n) * D3 + sbase) = pk;
    }
  }
}

// ---------------------------------------------------------------------------
// 2x2x2 max pool (spatial only), f16 -> f16
// ---------------------------------------------------------------------------
template <int C, int DIN>
__global__ void maxpool2(const _Float16* __restrict__ in, _Float16* __restrict__ out) {
  constexpr int DO = DIN / 2;
  long total = (long)NBATCH * C * DO * DO * DO;
  long id = (long)blockIdx.x * blockDim.x + threadIdx.x;
  if (id >= total) return;
  int x = (int)(id % DO); long r = id / DO;
  int y = (int)(r % DO);  r /= DO;
  int z = (int)(r % DO);  r /= DO;
  int c = (int)(r % C);
  int b = (int)(r / C);
  const _Float16* p = in + (((long)b * C + c) * DIN + 2 * z) * (DIN * DIN) + (2 * y) * DIN + 2 * x;
  float m = -1e30f;
#pragma unroll
  for (int dz = 0; dz < 2; ++dz)
#pragma unroll
    for (int dy = 0; dy < 2; ++dy)
#pragma unroll
      for (int dx = 0; dx < 2; ++dx) {
        float v = (float)p[dz * DIN * DIN + dy * DIN + dx];
        m = v > m ? v : m;
      }
  out[id] = (_Float16)m;
}

// ---------------------------------------------------------------------------
// Final linear 16384 -> 128 (tiny M=4): one wave per (b, o) dot product,
// vectorized 8 k per lane (v8h act + 2x float4 weights per iteration)
// ---------------------------------------------------------------------------
__global__ __launch_bounds__(256)
void final_linear(const _Float16* __restrict__ act,  // [B][16384] f16
                  const float* __restrict__ w,       // [128][16384]
                  const float* __restrict__ bias,    // [128]
                  float* __restrict__ y)             // [B][128]
{
  int gw = (int)((blockIdx.x * blockDim.x + threadIdx.x) >> 5);  // 0..511
  int lane = threadIdx.x & 31;
  int b = gw >> 7;
  int o = gw & 127;
  const _Float16* ap = act + (long)b * 16384;
  const float*    wp = w + (long)o * 16384;
  float acc = 0.0f;
  for (int kb = lane * 8; kb < 16384; kb += 32 * 8) {
    v8h av = *(const v8h*)(ap + kb);
    float4 w0 = *(const float4*)(wp + kb);
    float4 w1 = *(const float4*)(wp + kb + 4);
    acc += (float)av[0] * w0.x + (float)av[1] * w0.y +
           (float)av[2] * w0.z + (float)av[3] * w0.w +
           (float)av[4] * w1.x + (float)av[5] * w1.y +
           (float)av[6] * w1.z + (float)av[7] * w1.w;
  }
#pragma unroll
  for (int off = 16; off > 0; off >>= 1)
    acc += __shfl_xor(acc, off, 32);
  if (lane == 0) y[b * 128 + o] = acc + bias[o];
}

// ---------------------------------------------------------------------------
// BatchNorm over batch dim (B=4), per channel
// ---------------------------------------------------------------------------
__global__ void batchnorm(const float* __restrict__ y, const float* __restrict__ gamma,
                          const float* __restrict__ beta, float* __restrict__ out) {
  int c = threadIdx.x;
  if (c >= 128) return;
  float v0 = y[c], v1 = y[128 + c], v2 = y[256 + c], v3 = y[384 + c];
  float mu = 0.25f * (v0 + v1 + v2 + v3);
  float d0 = v0 - mu, d1 = v1 - mu, d2 = v2 - mu, d3 = v3 - mu;
  float var = 0.25f * (d0 * d0 + d1 * d1 + d2 * d2 + d3 * d3);
  float inv = rsqrtf(var + 1e-5f);
  float g = gamma[c], bt = beta[c];
  out[c]       = g * d0 * inv + bt;
  out[128 + c] = g * d1 * inv + bt;
  out[256 + c] = g * d2 * inv + bt;
  out[384 + c] = g * d3 * inv + bt;
}

// ---------------------------------------------------------------------------
extern "C" void kernel_launch(void* const* d_in, const int* in_sizes, int n_in,
                              void* d_out, int out_size, void* d_ws, size_t ws_size,
                              hipStream_t stream) {
  (void)in_sizes; (void)n_in; (void)out_size; (void)ws_size;

  const float* scene_pc = (const float*)d_in[0];
  const float* pt_w0    = (const float*)d_in[1];
  const float* pt_b0    = (const float*)d_in[2];
  const float* pt_w1    = (const float*)d_in[3];
  const float* pt_b1    = (const float*)d_in[4];
  const float* conv_w[4] = { (const float*)d_in[5], (const float*)d_in[7],
                             (const float*)d_in[9], (const float*)d_in[11] };
  const float* conv_b[4] = { (const float*)d_in[6], (const float*)d_in[8],
                             (const float*)d_in[10], (const float*)d_in[12] };
  const float* lin_w  = (const float*)d_in[13];
  const float* lin_b  = (const float*)d_in[14];
  const float* bn_g   = (const float*)d_in[15];
  const float* bn_b   = (const float*)d_in[16];
  float* out = (float*)d_out;

  // --- workspace layout ---
  char* ws = (char*)d_ws;
  size_t off = 0;
  auto carve = [&](size_t bytes) -> char* {
    char* p = ws + off;
    off += (bytes + 255) & ~(size_t)255;
    return p;
  };
  const long wsz[4] = { (long)512 * 256 * 27, (long)1024 * 512 * 27,
                        (long)2048 * 1024 * 27, (long)2048 * 2048 * 27 };
  _Float16* wh[4];
  for (int i = 0; i < 4; ++i) wh[i] = (_Float16*)carve((size_t)wsz[i] * 2);
  unsigned int* voxbuf = (unsigned int*)carve((size_t)NBATCH * 256 * NVOX * 4);
  _Float16* act0 = (_Float16*)carve((size_t)NBATCH * 256 * NVOX * 2);
  _Float16* ctmp = (_Float16*)carve((size_t)NBATCH * 512 * NVOX * 2);   // largest full-res conv out
  _Float16* act1 = (_Float16*)carve((size_t)NBATCH * 512 * 4096 * 2);
  _Float16* act2 = (_Float16*)carve((size_t)NBATCH * 1024 * 512 * 2);
  _Float16* act3 = (_Float16*)carve((size_t)NBATCH * 2048 * 64 * 2);
  _Float16* act4 = (_Float16*)carve((size_t)NBATCH * 2048 * 8 * 2);
  float* ybuf = (float*)carve(512 * 4);

  // 1) zero scatter buffer
  {
    long n = (long)NBATCH * 256 * NVOX;
    zero_u32<<<dim3((unsigned)((n + 255) / 256)), dim3(256), 0, stream>>>(voxbuf, n);
  }
  // 2) convert conv weights to f16 (same flat layout: [cout][cin*27])
  for (int i = 0; i < 4; ++i)
    f32_to_f16<<<dim3(4096), dim3(256), 0, stream>>>(conv_w[i], wh[i], wsz[i]);

  // 3) point MLP + scatter max (writes channel-major vox[b][c][voxel])
  point_mlp_scatter<<<dim3((NBATCH * NPTS) / 128), dim3(256), 0, stream>>>(
      scene_pc, pt_w0, pt_b0, pt_w1, pt_b1, voxbuf);

  // 4) bits -> f16 (already NCDHW; fully coalesced)
  {
    long n = (long)NBATCH * 256 * NVOX;
    vox_to_act<<<dim3((unsigned)((n + 255) / 256)), dim3(256), 0, stream>>>(voxbuf, act0);
  }

  // 5) conv + pool stack  (block tile: 32 M x 256 N, K-step 64, double-buffered)
  conv3d_wmma<256, 512, 32><<<dim3((NBATCH * 32768) / 32, 512 / 256), dim3(256), 0, stream>>>(
      act0, wh[0], conv_b[0], ctmp);
  {
    long n = (long)NBATCH * 512 * 4096;
    maxpool2<512, 32><<<dim3((unsigned)((n + 255) / 256)), dim3(256), 0, stream>>>(ctmp, act1);
  }
  conv3d_wmma<512, 1024, 16><<<dim3((NBATCH * 4096) / 32, 1024 / 256), dim3(256), 0, stream>>>(
      act1, wh[1], conv_b[1], ctmp);
  {
    long n = (long)NBATCH * 1024 * 512;
    maxpool2<1024, 16><<<dim3((unsigned)((n + 255) / 256)), dim3(256), 0, stream>>>(ctmp, act2);
  }
  conv3d_wmma<1024, 2048, 8><<<dim3((NBATCH * 512) / 32, 2048 / 256), dim3(256), 0, stream>>>(
      act2, wh[2], conv_b[2], ctmp);
  {
    long n = (long)NBATCH * 2048 * 64;
    maxpool2<2048, 8><<<dim3((unsigned)((n + 255) / 256)), dim3(256), 0, stream>>>(ctmp, act3);
  }
  conv3d_wmma<2048, 2048, 4><<<dim3((NBATCH * 64) / 32, 2048 / 256), dim3(256), 0, stream>>>(
      act3, wh[3], conv_b[3], ctmp);
  {
    long n = (long)NBATCH * 2048 * 8;
    maxpool2<2048, 4><<<dim3((unsigned)((n + 255) / 256)), dim3(256), 0, stream>>>(ctmp, act4);
  }

  // 6) final linear (one wave per output scalar: 512 waves)
  final_linear<<<dim3(64), dim3(256), 0, stream>>>(act4, lin_w, lin_b, ybuf);

  // 7) batchnorm over batch -> d_out
  batchnorm<<<dim3(1), dim3(128), 0, stream>>>(ybuf, bn_g, bn_b, out);
}
